// MCADBlock_75376676045417
// MI455X (gfx1250) — matrix operand order repaired and checked by reading, not compiled, not run on hardware
//
#include <hip/hip_runtime.h>

#define NN 4096
#define LL 200
#define HH 256
#define EE 131072
#define TT1 196
#define TT2 192

typedef __attribute__((ext_vector_type(16))) __bf16 v16bf;
typedef __attribute__((ext_vector_type(8)))  float  v8f;

union AB { v16bf v; unsigned u[8]; };

__device__ __forceinline__ unsigned short f2bf(float f) {
  union { float f; unsigned u; } x; x.f = f;
  unsigned r = x.u + 0x7FFFu + ((x.u >> 16) & 1u);
  return (unsigned short)(r >> 16);
}
__device__ __forceinline__ v8f wmma_bf16(v16bf a, v16bf b, v8f c) {
  return __builtin_amdgcn_wmma_f32_16x16x32_bf16(false, a, false, b, (short)0, c, false, false);
}
// load bf16 [p..p+7] and [p+16..p+23] (16B-aligned) into a fragment: two b128 loads
__device__ __forceinline__ void ld_frag16(AB& f, const unsigned short* p) {
  uint4 q0 = *(const uint4*)p;
  uint4 q1 = *(const uint4*)(p + 16);
  f.u[0] = q0.x; f.u[1] = q0.y; f.u[2] = q0.z; f.u[3] = q0.w;
  f.u[4] = q1.x; f.u[5] = q1.y; f.u[6] = q1.z; f.u[7] = q1.w;
}
// A-fragment: chunk-local k = {0..7}+8*hi (u0..3) and {16..23}+8*hi (u4..7)
__device__ __forceinline__ void ld_afrag(AB& f, const unsigned short* rowbase, int hi) {
  uint4 q0 = *(const uint4*)(rowbase + hi * 8);
  uint4 q1 = *(const uint4*)(rowbase + hi * 8 + 16);
  f.u[0] = q0.x; f.u[1] = q0.y; f.u[2] = q0.z; f.u[3] = q0.w;
  f.u[4] = q1.x; f.u[5] = q1.y; f.u[6] = q1.z; f.u[7] = q1.w;
}

// ---------------- weight prep: transpose + cvt to bf16 [o][K], pad rows ----
__global__ void prep_mat_kernel(const float* __restrict__ src, unsigned short* __restrict__ dst,
                                int K, int O, int Opad) {
  int i = blockIdx.x * 256 + threadIdx.x;
  if (i < Opad * K) {
    int o = i / K, k = i - o * K;
    dst[i] = (o < O) ? f2bf(src[k * O + o]) : (unsigned short)0;
  }
}
// tcn w2 [co][ci*5+kk] -> permuted bf16 [co][k''], k'' = ph*640 + kk*128 + cil
__global__ void prep_tcnw2_kernel(const float* __restrict__ w2, unsigned short* __restrict__ dst) {
  int i = blockIdx.x * 256 + threadIdx.x;
  if (i < 256 * 1280) {
    int co = i / 1280, kpp = i - co * 1280;
    int ph = kpp / 640, r = kpp - ph * 640;
    int kk = r >> 7, cil = r & 127;
    int ci = ph * 128 + cil;
    dst[i] = f2bf(w2[co * 1280 + ci * 5 + kk]);
  }
}
__global__ void cvt_bf16_kernel(const float* __restrict__ s, unsigned short* __restrict__ d, int n) {
  int i = blockIdx.x * 256 + threadIdx.x;
  if (i < n) d[i] = f2bf(s[i]);
}

// ---------------- TCN: conv1(relu) -> time-major bf16 LDS -> implicit-GEMM
// conv2 (WMMA, permuted-K) -> fused masked temporal mean. One block per node.
__global__ __launch_bounds__(256) void tcn_kernel(
    const float* __restrict__ x, const float* __restrict__ xmask,
    const float* __restrict__ b1, const float* __restrict__ b2,
    const float* __restrict__ w1, const unsigned short* __restrict__ w2p,
    float* __restrict__ outb)
{
  __shared__ float xs[LL];
  __shared__ float sm[TT2];
  __shared__ float msum_s;
  __shared__ unsigned short hA[TT1 * 128];   // [t][cil], half the channels per phase
  const int n = blockIdx.x;
  const int tid = threadIdx.x;
  if (tid < LL)  xs[tid] = x[n * LL + tid];
  if (tid < TT2) sm[tid] = xmask[n * LL + 4 + tid];
  __syncthreads();
  if (tid == 0) {
    float s = 0.f;
    for (int t = 0; t < TT2; ++t) s += sm[t];
    msum_s = (s != 0.f) ? s : 1.f;
  }
  const int lane = tid & 31, wv = tid >> 5;
  const int m = lane & 15, hi = lane >> 4;
  v8f acc[2][12] = {};
  for (int ph = 0; ph < 2; ++ph) {
    __syncthreads();
    { // conv1 + relu + bf16, channels [ph*128, ph*128+128), time-major store
      int cil = tid >> 1;
      int ci = ph * 128 + cil;
      float wk[5];
#pragma unroll
      for (int k = 0; k < 5; ++k) wk[k] = w1[ci * 5 + k];
      float bb = b1[ci];
      int t0 = (tid & 1) * 98;
      for (int t = t0; t < t0 + 98; ++t) {
        float a = bb;
#pragma unroll
        for (int k = 0; k < 5; ++k) a += xs[t + k] * wk[k];
        hA[t * 128 + cil] = f2bf(a > 0.f ? a : 0.f);
      }
    }
    __syncthreads();
    // permuted K: chunk kc -> kk = kc/4, cil base = (kc%4)*32
    for (int kc = 0; kc < 20; ++kc) {
      const int kk = kc >> 2;
      const int cb = (kc & 3) * 32;
      AB bu[2];
#pragma unroll
      for (int cti = 0; cti < 2; ++cti) {
        int co = (wv + cti * 8) * 16 + m;
        ld_frag16(bu[cti], w2p + co * 1280 + ph * 640 + kc * 32 + hi * 8);
      }
      if (kc < 19)
        __builtin_prefetch(w2p + (wv * 16 + m) * 1280 + ph * 640 + (kc + 1) * 32, 0, 0);
#pragma unroll
      for (int tt = 0; tt < 12; ++tt) {
        AB au;
        ld_afrag(au, hA + (tt * 16 + m + kk) * 128 + cb, hi);
        acc[0][tt] = wmma_bf16(au.v, bu[0].v, acc[0][tt]);
        acc[1][tt] = wmma_bf16(au.v, bu[1].v, acc[1][tt]);
      }
    }
  }
  // masked temporal mean; bias folds through normalization
#pragma unroll
  for (int cti = 0; cti < 2; ++cti) {
    int co = (wv + cti * 8) * 16 + m;
    float part = 0.f;
#pragma unroll
    for (int tt = 0; tt < 12; ++tt)
#pragma unroll
      for (int r = 0; r < 8; ++r)
        part += acc[cti][tt][r] * sm[tt * 16 + r + hi * 8];
    part += __shfl_xor(part, 16, 32);
    if (lane < 16) outb[n * HH + co] = part / msum_s + b2[co];
  }
}

// ---------------- Edge MLP (WMMA) + scatter-add aggregation ----------------
__global__ __launch_bounds__(256) void edge_kernel(
    const float* __restrict__ ea, const int* __restrict__ ei,
    const float* __restrict__ xin,
    const float* __restrict__ eb1, const float* __restrict__ eb2,
    const float* __restrict__ ew1, const unsigned short* __restrict__ ew2p,
    float* __restrict__ agg)
{
  const int lane = threadIdx.x & 31, wv = threadIdx.x >> 5;
  const int m = lane & 15, hi = lane >> 4;
  const int ebase = (blockIdx.x * 8 + wv) * 16;
  const float eav = ea[ebase + m];
  AB af[8];                       // hidden = relu(ea*w1 + b1), 16 edges x 256
#pragma unroll
  for (int kc = 0; kc < 8; ++kc)
#pragma unroll
    for (int v = 0; v < 8; ++v) {
      int k0 = kc * 32 + ((v < 4) ? (v * 2) : (16 + (v - 4) * 2)) + hi * 8;
      float h0 = eav * ew1[k0] + eb1[k0];         h0 = h0 > 0.f ? h0 : 0.f;
      float h1 = eav * ew1[k0 + 1] + eb1[k0 + 1]; h1 = h1 > 0.f ? h1 : 0.f;
      af[kc].u[v] = (unsigned)f2bf(h0) | ((unsigned)f2bf(h1) << 16);
    }
  int rIdx[8], cIdx[8];
#pragma unroll
  for (int r = 0; r < 8; ++r) {
    int e = ebase + r + hi * 8;
    rIdx[r] = ei[e];
    cIdx[r] = ei[EE + e];
  }
  for (int ct = 0; ct < 16; ++ct) {
    int co = ct * 16 + m;
    v8f acc = {};
#pragma unroll
    for (int kc = 0; kc < 8; ++kc) {
      AB bu;
      ld_frag16(bu, ew2p + co * HH + kc * 32 + hi * 8);
      acc = wmma_bf16(af[kc].v, bu.v, acc);
    }
    float bias = eb2[co];
#pragma unroll
    for (int r = 0; r < 8; ++r) {
      float msg = acc[r] + bias + xin[cIdx[r] * HH + co];
      atomicAdd(&agg[rIdx[r] * HH + co], msg);
    }
  }
}

// ---------------- Node MLP: cur = cur + mlp(concat(cur, agg)) (WMMA x2) ----
__global__ __launch_bounds__(256) void node_kernel(
    float* __restrict__ cur,
    const unsigned short* __restrict__ curb, const unsigned short* __restrict__ aggb,
    const float* __restrict__ nb1, const float* __restrict__ nb2,
    const unsigned short* __restrict__ nw1p, const unsigned short* __restrict__ nw2p)
{
  __shared__ unsigned short hid[8][16 * HH];
  const int lane = threadIdx.x & 31, wv = threadIdx.x >> 5;
  const int m = lane & 15, hi = lane >> 4;
  const int rbase = (blockIdx.x * 8 + wv) * 16;
  const int row = rbase + m;
  AB a1[16];                        // concat rows, K=512 (chunks 0-7 cur, 8-15 agg)
#pragma unroll
  for (int kc = 0; kc < 16; ++kc) {
    const unsigned short* base = (kc < 8) ? (curb + row * HH + kc * 32)
                                          : (aggb + row * HH + (kc - 8) * 32);
    ld_afrag(a1[kc], base, hi);
  }
  for (int ct = 0; ct < 16; ++ct) {
    int co = ct * 16 + m;
    if (ct < 15) __builtin_prefetch(nw1p + (co + 16) * 512, 0, 0);
    v8f acc = {};
#pragma unroll
    for (int kc = 0; kc < 16; ++kc) {
      AB bu;
      ld_frag16(bu, nw1p + co * 512 + kc * 32 + hi * 8);
      acc = wmma_bf16(a1[kc].v, bu.v, acc);
    }
    float bias = nb1[co];
#pragma unroll
    for (int r = 0; r < 8; ++r) {
      float h = acc[r] + bias;
      hid[wv][(r + hi * 8) * HH + co] = f2bf(h > 0.f ? h : 0.f);
    }
  }
  __syncthreads();
  AB a2[8];
#pragma unroll
  for (int kc = 0; kc < 8; ++kc)
    ld_afrag(a2[kc], &hid[wv][m * HH + kc * 32], hi);
  for (int ct = 0; ct < 16; ++ct) {
    int co = ct * 16 + m;
    v8f acc = {};
#pragma unroll
    for (int kc = 0; kc < 8; ++kc) {
      AB bu;
      ld_frag16(bu, nw2p + co * HH + kc * 32 + hi * 8);
      acc = wmma_bf16(a2[kc].v, bu.v, acc);
    }
    float bias = nb2[co];
#pragma unroll
    for (int r = 0; r < 8; ++r)
      cur[(rbase + r + hi * 8) * HH + co] += acc[r] + bias;
  }
}

// ---------------- Decoder: out(512) -> leaky_relu -> 200 (WMMA x2) ---------
__global__ __launch_bounds__(256) void dec_kernel(
    const unsigned short* __restrict__ outb,   // N x 512 bf16
    const float* __restrict__ db1, const float* __restrict__ db2,
    const unsigned short* __restrict__ dw1p,   // [256][512] bf16
    const unsigned short* __restrict__ dw2p,   // [208][256] bf16, zero-padded rows
    float* __restrict__ xhat)
{
  __shared__ unsigned short hid[8][16 * HH];
  const int lane = threadIdx.x & 31, wv = threadIdx.x >> 5;
  const int m = lane & 15, hi = lane >> 4;
  const int rbase = (blockIdx.x * 8 + wv) * 16;
  const int row = rbase + m;
  AB a1[16];
#pragma unroll
  for (int kc = 0; kc < 16; ++kc)
    ld_afrag(a1[kc], outb + row * 512 + kc * 32, hi);
  for (int ct = 0; ct < 16; ++ct) {
    int co = ct * 16 + m;
    v8f acc = {};
#pragma unroll
    for (int kc = 0; kc < 16; ++kc) {
      AB bu;
      ld_frag16(bu, dw1p + co * 512 + kc * 32 + hi * 8);
      acc = wmma_bf16(a1[kc].v, bu.v, acc);
    }
    float bias = db1[co];
#pragma unroll
    for (int r = 0; r < 8; ++r) {
      float h = acc[r] + bias;
      h = (h > 0.f) ? h : 0.01f * h;     // leaky relu
      hid[wv][(r + hi * 8) * HH + co] = f2bf(h);
    }
  }
  __syncthreads();
  AB a2[8];
#pragma unroll
  for (int kc = 0; kc < 8; ++kc)
    ld_afrag(a2[kc], &hid[wv][m * HH + kc * 32], hi);
  for (int ct = 0; ct < 13; ++ct) {      // 208 cols cover the 200 outputs
    int co = ct * 16 + m;
    v8f acc = {};
#pragma unroll
    for (int kc = 0; kc < 8; ++kc) {
      AB bu;
      ld_frag16(bu, dw2p + co * HH + kc * 32 + hi * 8);  // rows >=200 are zero
      acc = wmma_bf16(a2[kc].v, bu.v, acc);
    }
    if (co < LL) {
      float bias = db2[co];
#pragma unroll
      for (int r = 0; r < 8; ++r)
        xhat[(rbase + r + hi * 8) * LL + co] = acc[r] + bias;
    }
  }
}

// ---------------- Utility + context kernels --------------------------------
__global__ void zero_f_kernel(float* p, int n) {
  int i = blockIdx.x * 256 + threadIdx.x; if (i < n) p[i] = 0.f;
}
__global__ void zero_i_kernel(int* p, int n) {
  int i = blockIdx.x * 256 + threadIdx.x; if (i < n) p[i] = 0;
}
__global__ void concat_kernel(const float* __restrict__ a, const float* __restrict__ b,
                              float* __restrict__ o) {
  int i = blockIdx.x * 256 + threadIdx.x;
  int n = i >> 9, c = i & 511;
  o[i] = (c < HH) ? a[n * HH + c] : b[n * HH + c - HH];
}
__global__ void count_kernel(const int* __restrict__ ei, int* __restrict__ counts) {
  int e = blockIdx.x * 256 + threadIdx.x;
  if (e < EE) atomicAdd(&counts[ei[e]], 1);
}
__global__ void scan_kernel(const int* __restrict__ counts, int* __restrict__ offs,
                            int* __restrict__ cursor) {
  if (threadIdx.x == 0 && blockIdx.x == 0) {
    int run = 0;
    for (int i = 0; i < NN; ++i) { offs[i] = run; cursor[i] = run; run += counts[i]; }
    offs[NN] = run;
  }
}
__global__ void scatter_kernel(const int* __restrict__ ei, int* __restrict__ cursor,
                               int* __restrict__ colidx) {
  int e = blockIdx.x * 256 + threadIdx.x;
  if (e < EE) {
    int pos = atomicAdd(&cursor[ei[e]], 1);
    colidx[pos] = ei[EE + e];
  }
}
// one block per row: dedupe neighbors via LDS bitmap, unique-count-normalized mean
__global__ __launch_bounds__(256) void ctx_kernel(
    const int* __restrict__ offs, const int* __restrict__ colidx,
    const float* __restrict__ outc, float* __restrict__ res)
{
  __shared__ unsigned bm[128];
  __shared__ int cnt_s;
  const int i = blockIdx.x, tid = threadIdx.x;
  if (tid < 128) bm[tid] = 0u;
  if (tid == 0) cnt_s = 0;
  __syncthreads();
  const int s = offs[i], e2 = offs[i + 1];
  for (int e = s + tid; e < e2; e += 256) {
    int j = colidx[e];
    atomicOr(&bm[j >> 5], 1u << (j & 31));
  }
  __syncthreads();
  if (tid < 128) {
    int c = __popc(bm[tid]);
    if (c) atomicAdd(&cnt_s, c);
  }
  __syncthreads();
  float acc0 = 0.f, acc1 = 0.f;
  for (int w = 0; w < 128; ++w) {
    unsigned bits = bm[w];
    while (bits) {
      int b = __ffs(bits) - 1;
      bits &= bits - 1;
      int j = w * 32 + b;
      acc0 += outc[j * 512 + tid];
      acc1 += outc[j * 512 + tid + 256];
    }
  }
  int cnt = cnt_s;
  float inv = cnt > 0 ? 1.f / (float)cnt : 0.f;
  res[i * 512 + tid]       = outc[i * 512 + tid]       - acc0 * inv;
  res[i * 512 + tid + 256] = outc[i * 512 + tid + 256] - acc1 * inv;
}

// ---------------- Host orchestration ---------------------------------------
struct MlpP  { const float *b1, *b2, *w1, *w2; };
struct LayerP { MlpP edge, net; };

extern "C" void kernel_launch(void* const* d_in, const int* in_sizes, int n_in,
                              void* d_out, int out_size, void* d_ws, size_t ws_size,
                              hipStream_t stream)
{
  MlpP dec, tcn;
  LayerP semL[2], tmpL[2];
  const float *sem_x, *sem_ea, *tmp_x, *tmp_ea, *xmask;
  const int *sem_ei, *tmp_ei;

  if (n_in >= 47) {
    // params passed as 40 individual leaves in tree-flatten (alphabetical) order
    int i = 0;
    dec.b1 = (const float*)d_in[i++]; dec.b2 = (const float*)d_in[i++];
    dec.w1 = (const float*)d_in[i++]; dec.w2 = (const float*)d_in[i++];
    for (int l = 0; l < 2; ++l) {
      semL[l].edge.b1 = (const float*)d_in[i++]; semL[l].edge.b2 = (const float*)d_in[i++];
      semL[l].edge.w1 = (const float*)d_in[i++]; semL[l].edge.w2 = (const float*)d_in[i++];
      semL[l].net.b1  = (const float*)d_in[i++]; semL[l].net.b2  = (const float*)d_in[i++];
      semL[l].net.w1  = (const float*)d_in[i++]; semL[l].net.w2  = (const float*)d_in[i++];
    }
    tcn.b1 = (const float*)d_in[i++]; tcn.b2 = (const float*)d_in[i++];
    tcn.w1 = (const float*)d_in[i++]; tcn.w2 = (const float*)d_in[i++];
    for (int l = 0; l < 2; ++l) {
      tmpL[l].edge.b1 = (const float*)d_in[i++]; tmpL[l].edge.b2 = (const float*)d_in[i++];
      tmpL[l].edge.w1 = (const float*)d_in[i++]; tmpL[l].edge.w2 = (const float*)d_in[i++];
      tmpL[l].net.b1  = (const float*)d_in[i++]; tmpL[l].net.b2  = (const float*)d_in[i++];
      tmpL[l].net.w1  = (const float*)d_in[i++]; tmpL[l].net.w2  = (const float*)d_in[i++];
    }
    sem_x = (const float*)d_in[i++]; sem_ei = (const int*)d_in[i++]; sem_ea = (const float*)d_in[i++];
    tmp_x = (const float*)d_in[i++]; tmp_ei = (const int*)d_in[i++]; tmp_ea = (const float*)d_in[i++];
    xmask = (const float*)d_in[i++];
  } else {
    // params as one flat buffer, tree-flatten (alphabetical, recursive) order
    const float* p = (const float*)d_in[0];
    const long LSZ = 263424;
    dec = { p + 0, p + 256, p + 456, p + 131528 };
    for (int l = 0; l < 2; ++l) {
      long o = 182728 + l * LSZ;
      semL[l].edge = { p + o, p + o + 256, p + o + 512, p + o + 768 };
      semL[l].net  = { p + o + 66304, p + o + 66560, p + o + 66816, p + o + 197888 };
    }
    tcn = { p + 709576, p + 709832, p + 710088, p + 711368 };
    for (int l = 0; l < 2; ++l) {
      long o = 1039048 + l * LSZ;
      tmpL[l].edge = { p + o, p + o + 256, p + o + 512, p + o + 768 };
      tmpL[l].net  = { p + o + 66304, p + o + 66560, p + o + 66816, p + o + 197888 };
    }
    sem_x = (const float*)d_in[1]; sem_ei = (const int*)d_in[2]; sem_ea = (const float*)d_in[3];
    tmp_x = (const float*)d_in[4]; tmp_ei = (const int*)d_in[5]; tmp_ea = (const float*)d_in[6];
    xmask = (const float*)d_in[7];
  }

  // workspace carve-up (256B-aligned blocks)
  char* wsb = (char*)d_ws;
  auto alloc = [&](size_t bytes) -> char* {
    char* p = wsb; wsb += (bytes + 255) & ~(size_t)255; return p;
  };
  float* curS = (float*)alloc((size_t)NN * HH * 4);
  float* curT = (float*)alloc((size_t)NN * HH * 4);
  float* agg  = (float*)alloc((size_t)NN * HH * 4);
  unsigned short* curb = (unsigned short*)alloc((size_t)NN * HH * 2);
  unsigned short* aggb = (unsigned short*)alloc((size_t)NN * HH * 2);
  unsigned short* outbb = (unsigned short*)alloc((size_t)NN * 512 * 2);
  unsigned short* tcnw2p = (unsigned short*)alloc((size_t)256 * 1280 * 2);
  unsigned short *ew2p[2][2], *nw1p[2][2], *nw2p[2][2];
  for (int b = 0; b < 2; ++b)
    for (int l = 0; l < 2; ++l) {
      ew2p[b][l] = (unsigned short*)alloc((size_t)256 * 256 * 2);
      nw1p[b][l] = (unsigned short*)alloc((size_t)256 * 512 * 2);
      nw2p[b][l] = (unsigned short*)alloc((size_t)256 * 256 * 2);
    }
  unsigned short* dw1p = (unsigned short*)alloc((size_t)256 * 512 * 2);
  unsigned short* dw2p = (unsigned short*)alloc((size_t)208 * 256 * 2);
  int* counts = (int*)alloc((size_t)NN * 4);
  int* offs   = (int*)alloc((size_t)(NN + 1) * 4);
  int* cursor = (int*)alloc((size_t)NN * 4);
  int* colidx = (int*)alloc((size_t)EE * 4);

  // ---- one-time weight prep (bf16, [out][K], L2-resident afterwards) ----
  prep_tcnw2_kernel<<<(256 * 1280) / 256, 256, 0, stream>>>(tcn.w2, tcnw2p);
  const LayerP* Ls[2] = { semL, tmpL };
  for (int b = 0; b < 2; ++b)
    for (int l = 0; l < 2; ++l) {
      prep_mat_kernel<<<(256 * 256) / 256, 256, 0, stream>>>(Ls[b][l].edge.w2, ew2p[b][l], 256, 256, 256);
      prep_mat_kernel<<<(256 * 512) / 256, 256, 0, stream>>>(Ls[b][l].net.w1, nw1p[b][l], 512, 256, 256);
      prep_mat_kernel<<<(256 * 256) / 256, 256, 0, stream>>>(Ls[b][l].net.w2, nw2p[b][l], 256, 256, 256);
    }
  prep_mat_kernel<<<(256 * 512) / 256, 256, 0, stream>>>(dec.w1, dw1p, 512, 256, 256);
  prep_mat_kernel<<<(208 * 256) / 256, 256, 0, stream>>>(dec.w2, dw2p, 256, 200, 208);

  // ---- TCN (weights shared across branches) ----
  tcn_kernel<<<NN, 256, 0, stream>>>(sem_x, xmask, tcn.b1, tcn.b2, tcn.w1, tcnw2p, curS);
  tcn_kernel<<<NN, 256, 0, stream>>>(tmp_x, xmask, tcn.b1, tcn.b2, tcn.w1, tcnw2p, curT);

  // ---- GNN layers ----
  float* curs[2] = { curS, curT };
  const float* eas[2] = { sem_ea, tmp_ea };
  const int* eis[2] = { sem_ei, tmp_ei };
  for (int b = 0; b < 2; ++b)
    for (int l = 0; l < 2; ++l) {
      zero_f_kernel<<<(NN * HH) / 256, 256, 0, stream>>>(agg, NN * HH);
      edge_kernel<<<EE / 128, 256, 0, stream>>>(eas[b], eis[b], curs[b],
          Ls[b][l].edge.b1, Ls[b][l].edge.b2, Ls[b][l].edge.w1, ew2p[b][l], agg);
      cvt_bf16_kernel<<<(NN * HH) / 256, 256, 0, stream>>>(curs[b], curb, NN * HH);
      cvt_bf16_kernel<<<(NN * HH) / 256, 256, 0, stream>>>(agg, aggb, NN * HH);
      node_kernel<<<NN / 128, 256, 0, stream>>>(curs[b], curb, aggb,
          Ls[b][l].net.b1, Ls[b][l].net.b2, nw1p[b][l], nw2p[b][l]);
    }

  // ---- out = concat(out1, out2) -> section 0; bf16 copy for decoder ----
  float* out = (float*)d_out;
  concat_kernel<<<(NN * 512) / 256, 256, 0, stream>>>(curS, curT, out);
  cvt_bf16_kernel<<<(NN * 512) / 256, 256, 0, stream>>>(out, outbb, NN * 512);

  // ---- x_hat -> section 3 ----
  dec_kernel<<<NN / 128, 256, 0, stream>>>(outbb, dec.b1, dec.b2, dw1p, dw2p,
                                           out + (size_t)3 * NN * 512);

  // ---- contexts: CSR build then dedupe-bitmap aggregation -> sections 1,2 ----
  for (int c = 0; c < 2; ++c) {
    zero_i_kernel<<<NN / 256, 256, 0, stream>>>(counts, NN);
    count_kernel<<<EE / 256, 256, 0, stream>>>(eis[c], counts);
    scan_kernel<<<1, 64, 0, stream>>>(counts, offs, cursor);
    scatter_kernel<<<EE / 256, 256, 0, stream>>>(eis[c], cursor, colidx);
    ctx_kernel<<<NN, 256, 0, stream>>>(offs, colidx, out,
                                       out + (size_t)(c + 1) * NN * 512);
  }
}
